// GeometricSemanticInterface_21629455303272
// MI455X (gfx1250) — matrix operand order repaired and checked by reading, not compiled, not run on hardware
//
#include <hip/hip_runtime.h>

typedef float    v8f  __attribute__((ext_vector_type(8)));
typedef float    v2f  __attribute__((ext_vector_type(2)));
typedef _Float16 v16h __attribute__((ext_vector_type(16)));
typedef _Float16 v8h  __attribute__((ext_vector_type(8)));

#define NW   8     // waves per block
#define RPW  32    // rows per wave
#define RPB  256   // rows per block

__device__ __forceinline__ float sigmoidf_(float x) { return 1.0f / (1.0f + __expf(-x)); }
__device__ __forceinline__ float tanhf_(float x) {
  float t = __expf(2.0f * x);
  return (t - 1.0f) / (t + 1.0f);
}

__global__ __launch_bounds__(256, 2)
void gsi_fused(const float* __restrict__ states,
               const float* __restrict__ w1, const float* __restrict__ b1,
               const float* __restrict__ w2, const float* __restrict__ b2,
               const float* __restrict__ w3, const float* __restrict__ b3,
               const float* __restrict__ pw, const float* __restrict__ pb,
               const float* __restrict__ fwh, const float* __restrict__ fbh,
               const float* __restrict__ fwmu, const float* __restrict__ fbmu,
               const float* __restrict__ fwa, const float* __restrict__ fba,
               float* __restrict__ out, int nrows, int nchunks)
{
  // per-wave activation staging (intra-wave traffic only)
  __shared__ __align__(16) _Float16 sh1[NW][32][64];   // 32 KB
  __shared__ __align__(16) _Float16 sh2[NW][32][32];   // 16 KB
  __shared__ __align__(16) float    scc[NW][32][8];    //  8 KB

  const int lane = threadIdx.x & 31;
  const int wid  = threadIdx.x >> 5;
  const int half = lane >> 4;   // 0: lanes 0-15, 1: lanes 16-31
  const int lm   = lane & 15;

  // ---------------- weight fragments (built once per block) ----------------
  // layer-1 B fragments: f32 4x16 tiles. lanes0-15 hold K=0,1; lanes16-31 K=2,3
  v2f w1f[4]; float bias1[4];
  #pragma unroll
  for (int nt = 0; nt < 4; ++nt) {
    const int n  = nt * 16 + lm;
    const int kb = half * 2;
    w1f[nt][0] = w1[(kb + 0) * 64 + n];
    w1f[nt][1] = w1[(kb + 1) * 64 + n];
    bias1[nt]  = b1[n];
  }
  // layer-2 B fragments: f16 32x16 tiles, lane holds K 8h..8h+7 and 16+8h..16+8h+7 of one column
  v16h w2f[2][2]; float bias2[2];
  #pragma unroll
  for (int nt = 0; nt < 2; ++nt) {
    const int n = nt * 16 + lm;
    bias2[nt] = b2[n];
    #pragma unroll
    for (int kc = 0; kc < 2; ++kc) {
      const int kb = kc * 32 + half * 8;
      #pragma unroll
      for (int e = 0; e < 16; ++e)
        w2f[nt][kc][e] = (_Float16)w2[(kb + (e & 7) + ((e >> 3) << 4)) * 32 + n];
    }
  }
  // layer-3 B fragment: 32x8 padded to 32x16 with zeros
  v16h w3f; float bias3 = (lm < 8) ? b3[lm] : 0.0f;
  {
    const int kb = half * 8;
    #pragma unroll
    for (int e = 0; e < 16; ++e)
      w3f[e] = (lm < 8) ? (_Float16)w3[(kb + (e & 7) + ((e >> 3) << 4)) * 8 + lm] : (_Float16)0.0f;
  }

  for (int chunk = blockIdx.x; chunk < nchunks; chunk += (int)gridDim.x) {
    const long wbase = (long)chunk * RPB + wid * RPW;

    #pragma unroll
    for (int t = 0; t < 2; ++t) {
      // ---- layer 1: states(16x4) @ w1(4x64) via V_WMMA_F32_16X16X4_F32 ----
      long arow = wbase + t * 16 + lm;           // A: lane holds row M=lm, K-pair by half
      if (arow > (long)nrows - 1) arow = (long)nrows - 1;
      v2f a1;
      a1[0] = states[arow * 4 + half * 2 + 0];
      a1[1] = states[arow * 4 + half * 2 + 1];
      #pragma unroll
      for (int nt = 0; nt < 4; ++nt) {
        v8f c = {};
        c = __builtin_amdgcn_wmma_f32_16x16x4_f32(false, a1, false, w1f[nt],
                                                  (short)0, c, false, false);
        #pragma unroll
        for (int v = 0; v < 8; ++v) {           // D: lane col N=lm, rows v(+8*half)
          float hv = fmaxf(c[v] + bias1[nt], 0.0f);
          sh1[wid][t * 16 + half * 8 + v][nt * 16 + lm] = (_Float16)hv;
        }
      }
      __builtin_amdgcn_wave_barrier();

      // ---- layer 2: h1(16x64) @ w2(64x32) via V_WMMA_F32_16X16X32_F16 ----
      const int r = t * 16 + lm;
      #pragma unroll
      for (int nt = 0; nt < 2; ++nt) {
        v8f c = {};
        #pragma unroll
        for (int kc = 0; kc < 2; ++kc) {
          const int kb = kc * 32 + half * 8;    // A 16-bit layout: lanes0-15 K 0-7/16-23
          v8h lo = *(const v8h*)&sh1[wid][r][kb];
          v8h hi = *(const v8h*)&sh1[wid][r][kb + 16];
          v16h a = __builtin_shufflevector(lo, hi,
                     0,1,2,3,4,5,6,7,8,9,10,11,12,13,14,15);
          c = __builtin_amdgcn_wmma_f32_16x16x32_f16(false, a, false, w2f[nt][kc],
                                                     (short)0, c, false, false);
        }
        #pragma unroll
        for (int v = 0; v < 8; ++v) {
          float hv = fmaxf(c[v] + bias2[nt], 0.0f);
          sh2[wid][t * 16 + half * 8 + v][nt * 16 + lm] = (_Float16)hv;
        }
      }
      __builtin_amdgcn_wave_barrier();

      // ---- layer 3: h2(16x32) @ w3(32x8 zero-padded to 16) ----
      {
        const int kb = half * 8;
        v8h lo = *(const v8h*)&sh2[wid][r][kb];
        v8h hi = *(const v8h*)&sh2[wid][r][kb + 16];
        v16h a = __builtin_shufflevector(lo, hi,
                   0,1,2,3,4,5,6,7,8,9,10,11,12,13,14,15);
        v8f c = {};
        c = __builtin_amdgcn_wmma_f32_16x16x32_f16(false, a, false, w3f,
                                                   (short)0, c, false, false);
        if (lm < 8) {
          #pragma unroll
          for (int v = 0; v < 8; ++v)
            scc[wid][t * 16 + half * 8 + v][lm] = c[v] + bias3;
        }
      }
      __builtin_amdgcn_wave_barrier();
    }

    // -------- per-lane scalar tail: projection + masked MAF flow --------
    const long row = wbase + lane;               // lane <-> row, 1:1
    float cc[8];
    #pragma unroll
    for (int k = 0; k < 8; ++k) cc[k] = scc[wid][lane][k];

    float x0 = pb[0], x1 = pb[1], x2 = pb[2];
    #pragma unroll
    for (int k = 0; k < 8; ++k) {
      x0 = fmaf(cc[k], pw[k * 3 + 0], x0);
      x1 = fmaf(cc[k], pw[k * 3 + 1], x1);
      x2 = fmaf(cc[k], pw[k * 3 + 2], x2);
    }

    // MADE mask structure hardcoded:
    //   m_h: row0 all-1, row1 = (j odd), row2 = 0
    //   m_o: col0 = 0, col1 = (j even), col2 all-1
    for (int l = 0; l < 10; ++l) {
      const float* wh  = fwh  + l * 96;
      const float* bh  = fbh  + l * 32;
      const float* wmu = fwmu + l * 96;
      const float* wa  = fwa  + l * 96;
      float mu0 = fbmu[l * 3 + 0];
      float mu1 = fbmu[l * 3 + 1];
      float mu2 = fbmu[l * 3 + 2];
      float al0 = fba[l * 3 + 0];
      float al1 = fba[l * 3 + 1];
      float al2 = fba[l * 3 + 2];
      #pragma unroll
      for (int j = 0; j < 32; j += 2) {
        // even hidden unit: depends on x0 only; feeds mu1/al1 and mu2/al2
        float he = fmaxf(fmaf(x0, wh[j], bh[j]), 0.0f);
        mu1 = fmaf(he, wmu[j * 3 + 1], mu1);
        al1 = fmaf(he, wa [j * 3 + 1], al1);
        mu2 = fmaf(he, wmu[j * 3 + 2], mu2);
        al2 = fmaf(he, wa [j * 3 + 2], al2);
        // odd hidden unit: depends on x0,x1; feeds mu2/al2 only
        float ho = fmaxf(fmaf(x1, wh[32 + j + 1], fmaf(x0, wh[j + 1], bh[j + 1])), 0.0f);
        mu2 = fmaf(ho, wmu[(j + 1) * 3 + 2], mu2);
        al2 = fmaf(ho, wa [(j + 1) * 3 + 2], al2);
      }
      float nx0 = (x0 - mu0) * __expf(-tanhf_(al0));
      float nx1 = (x1 - mu1) * __expf(-tanhf_(al1));
      float nx2 = (x2 - mu2) * __expf(-tanhf_(al2));
      x0 = nx2; x1 = nx1; x2 = nx0;              // jnp.flip
    }

    if (row < (long)nrows) {
      out[row * 3 + 0] = 6.283185307179586f * sigmoidf_(x0);
      out[row * 3 + 1] = sigmoidf_(x1);
      out[row * 3 + 2] = sigmoidf_(x2);
    }
    __builtin_amdgcn_wave_barrier();
  }
}

extern "C" void kernel_launch(void* const* d_in, const int* in_sizes, int n_in,
                              void* d_out, int out_size, void* d_ws, size_t ws_size,
                              hipStream_t stream)
{
  const float* states = (const float*)d_in[0];
  const float* w1   = (const float*)d_in[1];
  const float* b1   = (const float*)d_in[2];
  const float* w2   = (const float*)d_in[3];
  const float* b2   = (const float*)d_in[4];
  const float* w3   = (const float*)d_in[5];
  const float* b3   = (const float*)d_in[6];
  const float* pw   = (const float*)d_in[7];
  const float* pb   = (const float*)d_in[8];
  const float* fwh  = (const float*)d_in[9];
  const float* fbh  = (const float*)d_in[10];
  const float* fwmu = (const float*)d_in[11];
  const float* fbmu = (const float*)d_in[12];
  const float* fwa  = (const float*)d_in[13];
  const float* fba  = (const float*)d_in[14];
  float* out = (float*)d_out;

  const int nrows   = in_sizes[0] / 4;
  const int nchunks = (nrows + RPB - 1) / RPB;
  int grid = nchunks < 2048 ? nchunks : 2048;
  if (grid < 1) grid = 1;

  gsi_fused<<<grid, 256, 0, stream>>>(states, w1, b1, w2, b2, w3, b3, pw, pb,
                                      fwh, fbh, fwmu, fbmu, fwa, fba,
                                      out, nrows, nchunks);
}
